// Generator_10428180594771
// MI455X (gfx1250) — matrix operand order repaired and checked by reading, not compiled, not run on hardware
//
#include <hip/hip_runtime.h>
#include <cmath>

#define LRELU_SLOPE 0.2f
#define SQRT2_F 1.41421356237309515f
#define RSQRT2_F 0.70710678118654752f

typedef unsigned short uh;
typedef unsigned int uint32;
typedef __attribute__((ext_vector_type(4)))  int    v4i;
typedef __attribute__((ext_vector_type(8)))  int    v8i;
typedef __attribute__((ext_vector_type(8)))  uh     v8us;
typedef __attribute__((ext_vector_type(16))) uh     v16us;
typedef __attribute__((ext_vector_type(16))) __bf16 v16bf;
typedef __attribute__((ext_vector_type(8)))  float  v8f;

__device__ __forceinline__ uh f2bf(float f) {
  unsigned u = __builtin_bit_cast(unsigned, f);
  u += 0x7FFFu + ((u >> 16) & 1u);          // round-to-nearest-even
  return (uh)(u >> 16);
}

__device__ __forceinline__ float lrelu_g(float v) {
  return (v > 0.f ? v : LRELU_SLOPE * v) * SQRT2_F;
}

// Issue a TDM 2D-tile load: 64 rows x 32 bf16 cols of W16[K-major] -> LDS.
// D# group0: count=1, lds_addr, global tile addr, type=2("image").
// D# group1: data_size=1(2B), tensor dims = remaining (OOB rows/cols zero-fill),
//            tile 32x64, dim0 stride = K.
__device__ __forceinline__ void tdm_load_b(
    const uh* gsrc, unsigned lds_addr, int tdim0, int tdim1, int K)
{
  unsigned long long gaddr = (unsigned long long)(const void*)gsrc;
  v4i g0;
  g0[0] = 1;
  g0[1] = (int)lds_addr;
  g0[2] = (int)(unsigned)(gaddr & 0xFFFFFFFFull);
  g0[3] = (int)(((unsigned)(gaddr >> 32) & 0x1FFFFFFu) | (2u << 30));
  v8i g1;
  g1[0] = 0x10000;                                   // wg_mask=0, data_size=1 (2B)
  g1[1] = (tdim0 & 0xFFFF) << 16;                    // tensor_dim0[15:0]
  g1[2] = (int)(((unsigned)tdim0 >> 16) | (((unsigned)tdim1 & 0xFFFFu) << 16));
  g1[3] = (int)(((unsigned)tdim1 >> 16) | (32u << 16)); // tile_dim0 = 32
  g1[4] = 64;                                        // tile_dim1 = 64
  g1[5] = K;                                         // tensor_dim0_stride[31:0]
  g1[6] = 0;
  g1[7] = 0;
  asm volatile("tensor_load_to_lds %0, %1" :: "s"(g0), "s"(g1) : "memory");
}

// ---------------------------------------------------------------------------
// Implicit-GEMM conv via V_WMMA_F32_16X16X32_BF16 + software-pipelined TDM.
//   C[M,N] = im2col(x)[M,K] * W16^T[K,N], M = Ho*Wo per image, N = Co, K = Ci*KH*KW
// Weights pre-converted to bf16 (scale folded). B tile (64x32 bf16) is fetched
// by TENSOR_LOAD_TO_LDS into a ping-pong buffer: the DMA for step i+1 is issued
// right after the loop-top barrier (that buffer was last read at step i-1), the
// wave then does the im2col A gather, and only waits TENSORcnt<=1 before use.
// MODE 0: normal conv (stride, pad). MODE 1: lhs_dilation=2, pad=2 (weights
// pre-flipped). Block = 256 thr = 8 waves; tile 128M x 64N x 32K; each wave
// computes a 16x64 strip (4 WMMAs / K-step).
// ---------------------------------------------------------------------------
template<int KH, int KW, int MODE>
__global__ __launch_bounds__(256) void k_conv_wmma(
    const float* __restrict__ x, const uh* __restrict__ w16,
    const float* __restrict__ bias, float* __restrict__ y,
    int Ci, int Co, int H, int W, int Ho, int Wo,
    int stride, int pad, int act, long long wbstride)
{
  __shared__ uh As[128][32];      // [pixel row][k]          8KB
  __shared__ uh Bs[2][64][32];    // ping-pong TDM dest      8KB

  const int KHKW = KH * KW;
  const int b  = blockIdx.z;
  const int M  = Ho * Wo;
  const int K  = Ci * KHKW;
  const int m0 = blockIdx.x * 128;
  const int n0 = blockIdx.y * 64;

  const float* xb = x + (long long)b * Ci * H * W;
  const uh*    wb = w16 + (long long)b * wbstride + (long long)n0 * K;
  float*       yb = y + (long long)b * Co * M;

  const int t    = threadIdx.x;
  const int frow = t >> 1;          // 0..127 : A row handled by this thread
  const int fk   = (t & 1) * 16;    // 0 / 16 : A k-offset

  const int lane  = t & 31;
  const int wid   = t >> 5;
  const int wm    = wid * 16;       // wave M offset in tile
  const int lrow  = lane & 15;
  const int lhalf = lane >> 4;

  const int am  = m0 + frow;
  const int aoy = am / Wo;
  const int aox = am - aoy * Wo;
  const bool arow_ok = am < M;

  const unsigned lds_b = (unsigned)(unsigned long long)&Bs[0][0][0];
  const int tdim1 = Co - n0;

  v8f acc[4] = {{}, {}, {}, {}};

  const int nk = (K + 31) >> 5;
  if (t < 32) tdm_load_b(wb, lds_b, K, tdim1, K);       // prefetch step 0

  for (int i = 0; i < nk; ++i) {
    const int k0 = i << 5;
    const int cur = i & 1;
    __syncthreads();   // all reads of Bs[cur^1] (step i-1) and As (step i-1) done
    if (t < 32 && i + 1 < nk)                           // prefetch step i+1
      tdm_load_b(wb + (k0 + 32), lds_b + (unsigned)((cur ^ 1) * 64 * 32 * 2),
                 K - (k0 + 32), tdim1, K);

    // ---- fill A tile (im2col gather, f32 -> bf16, paired 32-bit LDS stores) ----
#pragma unroll
    for (int j = 0; j < 16; j += 2) {
      uint32 pk = 0;
#pragma unroll
      for (int e = 0; e < 2; ++e) {
        int kk = k0 + fk + j + e;
        uh v = 0;
        if (arow_ok && kk < K) {
          int ci  = kk / KHKW;               // constant divisor -> mul/shift
          int rem = kk - ci * KHKW;
          int ky  = rem / KW;
          int kx  = rem - ky * KW;
          float xv = 0.f;
          if (MODE == 0) {
            int iy = aoy * stride - pad + ky;
            int ix = aox * stride - pad + kx;
            if (iy >= 0 && iy < H && ix >= 0 && ix < W)
              xv = xb[((long long)ci * H + iy) * W + ix];
          } else {  // lhs_dilation = 2, effective pad = KH-1 = 2
            int dy = aoy + ky - 2, dx = aox + kx - 2;
            if (dy >= 0 && dx >= 0 && !(dy & 1) && !(dx & 1)) {
              int iy = dy >> 1, ix = dx >> 1;
              if (iy < H && ix < W)
                xv = xb[((long long)ci * H + iy) * W + ix];
            }
          }
          v = f2bf(xv);
        }
        pk |= (uint32)v << (16 * e);
      }
      *(uint32*)&As[frow][fk + j] = pk;
    }

    // oldest outstanding TDM (current tile) must be complete; the prefetch may fly
    if (t < 32) {
      if (i + 1 < nk) __builtin_amdgcn_s_wait_tensorcnt(1);
      else            __builtin_amdgcn_s_wait_tensorcnt(0);
    }
    __syncthreads();

    // ---- A fragment: 16x32, lane<16 -> K 0-7 & 16-23, lane>=16 -> K 8-15 & 24-31
    v8us alo = *(const v8us*)&As[wm + lrow][lhalf * 8];
    v8us ahi = *(const v8us*)&As[wm + lrow][16 + lhalf * 8];
    v16bf aF = __builtin_bit_cast(v16bf, (v16us)__builtin_shufflevector(
        alo, ahi, 0,1,2,3,4,5,6,7,8,9,10,11,12,13,14,15));

    // ---- 4 B fragments (32x16 each): lane<16 -> K 0-7 & 16-23, lane>=16 -> K 8-15 & 24-31
#pragma unroll
    for (int g = 0; g < 4; ++g) {
      v8us blo = *(const v8us*)&Bs[cur][g * 16 + lrow][lhalf * 8];
      v8us bhi = *(const v8us*)&Bs[cur][g * 16 + lrow][16 + lhalf * 8];
      v16bf bF = __builtin_bit_cast(v16bf, (v16us)__builtin_shufflevector(
          blo, bhi, 0,1,2,3,4,5,6,7,8,9,10,11,12,13,14,15));
      acc[g] = __builtin_amdgcn_wmma_f32_16x16x32_bf16(false, aF, false, bF,
                                                       (short)0, acc[g], false, false);
    }
  }

  // ---- epilogue: C/D layout VGPR i -> M = i + 8*lhalf, N = lane&15
  const int crow = wm + lhalf * 8;
#pragma unroll
  for (int g = 0; g < 4; ++g) {
    int gn = n0 + g * 16 + lrow;
    if (gn >= Co) continue;
    float bval = bias ? bias[gn] : 0.f;
#pragma unroll
    for (int i = 0; i < 8; ++i) {
      int gm = m0 + crow + i;
      if (gm >= M) continue;
      float d = acc[g][i] + bval;
      if (act) d = lrelu_g(d);
      yb[(long long)gn * M + gm] = d;
    }
  }
}

// ---------------------------------------------------------------------------
// upfirdn2d with the fixed [1,3,3,1] outer-product kernel; up in {1,2}, down=1.
// ---------------------------------------------------------------------------
__global__ __launch_bounds__(256) void k_upfirdn(
    const float* __restrict__ x, float* __restrict__ y,
    int C, int H, int W, int Ho, int Wo,
    int up, int pad0, float gain, int addto)
{
  long long i = (long long)blockIdx.x * 256 + threadIdx.x;
  long long total = (long long)8 * C * Ho * Wo;
  if (i >= total) return;
  int ox = (int)(i % Wo); long long r = i / Wo;
  int oy = (int)(r % Ho); r /= Ho;
  int bc = (int)r;
  const float k1[4] = {0.125f, 0.375f, 0.375f, 0.125f};
  const int Dh = (H - 1) * up + 1;
  const int Dw = (W - 1) * up + 1;
  const float* xp = x + (long long)bc * H * W;
  float sum = 0.f;
#pragma unroll
  for (int ky = 0; ky < 4; ++ky) {
    int dy = oy - pad0 + ky;
    if (dy < 0 || dy >= Dh) continue;
    if (up == 2 && (dy & 1)) continue;
    int iy = (up == 2) ? (dy >> 1) : dy;
#pragma unroll
    for (int kx = 0; kx < 4; ++kx) {
      int dx = ox - pad0 + kx;
      if (dx < 0 || dx >= Dw) continue;
      if (up == 2 && (dx & 1)) continue;
      int ix = (up == 2) ? (dx >> 1) : dx;
      sum += k1[ky] * k1[kx] * xp[iy * W + ix];
    }
  }
  sum *= gain;
  long long o = (long long)bc * Ho * Wo + (long long)oy * Wo + ox;
  if (addto) y[o] += sum; else y[o] = sum;
}

// x = lrelu(x + nw*noise + bias[c]) * sqrt2, in place
__global__ __launch_bounds__(256) void k_noise_bias_act(
    float* __restrict__ x, const float* __restrict__ noise,
    const float* __restrict__ nw, const float* __restrict__ bias,
    int C, int H, int W)
{
  long long i = (long long)blockIdx.x * 256 + threadIdx.x;
  long long total = (long long)8 * C * H * W;
  if (i >= total) return;
  int hw = (int)(i % ((long long)H * W));
  int c  = (int)((i / ((long long)H * W)) % C);
  float v = x[i] + nw[0] * noise[hw] + bias[c];
  x[i] = lrelu_g(v);
}

__global__ __launch_bounds__(256) void k_combine(
    const float* __restrict__ a, const float* __restrict__ b,
    float* __restrict__ y, long long n)
{
  long long i = (long long)blockIdx.x * 256 + threadIdx.x;
  if (i < n) y[i] = (a[i] + b[i]) * RSQRT2_F;
}

// y[b][n] = act( scale * dot(x[b,:], w[n,:]) + bias[n] ), B = 8
__global__ __launch_bounds__(256) void k_linear(
    const float* __restrict__ x, const float* __restrict__ w,
    const float* __restrict__ bias, float* __restrict__ y,
    int K, int N, float scale, int act)
{
  int i = blockIdx.x * 256 + threadIdx.x;
  if (i >= 8 * N) return;
  int bb = i / N, n = i - bb * N;
  const float* xr = x + (long long)bb * K;
  const float* wr = w + (long long)n * K;
  float s = 0.f;
  for (int k = 0; k < K; ++k) s += xr[k] * wr[k];
  s = s * scale + bias[n];
  if (act) s = lrelu_g(s);
  y[i] = s;
}

// plain weight -> bf16 with scale folded
__global__ __launch_bounds__(256) void k_w2bf(
    const float* __restrict__ w, uh* __restrict__ o, long long n, float scale)
{
  long long i = (long long)blockIdx.x * 256 + threadIdx.x;
  if (i < n) o[i] = f2bf(w[i] * scale);
}

// per-(b,co) modulated weight -> bf16: wscale*w*s[ci] * (demod? rsqrt(sumsq+1e-8):1)
// optional spatial flip (for the lhs-dilated transposed conv)
__global__ __launch_bounds__(256) void k_modulate(
    const float* __restrict__ w, const float* __restrict__ s,
    uh* __restrict__ wout,
    int Co, int Ci, int kh, int kw, float wscale, int demod, int flip)
{
  __shared__ float red[256];
  int b  = blockIdx.x / Co;
  int co = blockIdx.x - b * Co;
  int khkw = kh * kw;
  int K = Ci * khkw;
  const float* wr = w + (long long)co * K;
  const float* sr = s + (long long)b * Ci;
  float psum = 0.f;
  for (int k = threadIdx.x; k < K; k += 256) {
    float v = wscale * wr[k] * sr[k / khkw];
    psum += v * v;
  }
  red[threadIdx.x] = psum;
  __syncthreads();
  for (int off = 128; off > 0; off >>= 1) {
    if ((int)threadIdx.x < off) red[threadIdx.x] += red[threadIdx.x + off];
    __syncthreads();
  }
  float d = demod ? rsqrtf(red[0] + 1e-8f) : 1.f;
  uh* wo = wout + ((long long)b * Co + co) * K;
  for (int k = threadIdx.x; k < K; k += 256) {
    int ci  = k / khkw;
    int rem = k - ci * khkw;
    float v = wscale * wr[k] * sr[ci] * d;
    int ok = k;
    if (flip) {
      int ky = rem / kw, kx = rem - ky * kw;
      ok = ci * khkw + (kh - 1 - ky) * kw + (kw - 1 - kx);
    }
    wo[ok] = f2bf(v);
  }
}

// ---------------------------------------------------------------------------
// Host orchestration of the full forward pass.
// ---------------------------------------------------------------------------
extern "C" void kernel_launch(void* const* d_in, const int* in_sizes, int n_in,
                              void* d_out, int out_size, void* d_ws, size_t ws_size,
                              hipStream_t stream)
{
  (void)in_sizes; (void)n_in; (void)out_size; (void)ws_size;
  const int B = 8;
  auto F = [&](int i) { return (const float*)d_in[i]; };

  // workspace layout (in floats; bf16 buffers use 2 uh per float slot)
  float* ws = (float*)d_ws;
  const size_t CAP = 67700000;           // >= 8*128*257*257
  float* B0 = ws;  ws += CAP;
  float* B1 = ws;  ws += CAP;
  float* B2 = ws;  ws += CAP;
  float* B3 = ws;  ws += CAP;
  uh* W16 = (uh*)ws; ws += 9500000;      // >= 8*512*512*9 bf16
  float* STY  = ws; ws += 8 * 512;
  float* HB   = ws; ws += 8 * 512;
  float* ZB   = ws; ws += 8 * 512;
  float* FEAT = ws; ws += 8 * 512 * 16;
  float* SK0  = ws; ws += (size_t)8 * 3 * 256 * 256;
  float* SK1  = ws; ws += (size_t)8 * 3 * 256 * 256;

  auto conv = [&](const float* x, const uh* w16, const float* bias, float* y,
                  int Ci, int Co, int H, int kk, int stride, int pad,
                  int mode, int act, long long wbstride) -> int {
    int Ho = (mode == 0) ? ((H + 2 * pad - kk) / stride + 1) : (2 * H + 1);
    int M = Ho * Ho;
    dim3 g((M + 127) / 128, (Co + 63) / 64, B);
    if (mode == 1)
      k_conv_wmma<3,3,1><<<g, 256, 0, stream>>>(x, w16, bias, y, Ci, Co, H, H,
                                                Ho, Ho, stride, pad, act, wbstride);
    else if (kk == 3)
      k_conv_wmma<3,3,0><<<g, 256, 0, stream>>>(x, w16, bias, y, Ci, Co, H, H,
                                                Ho, Ho, stride, pad, act, wbstride);
    else
      k_conv_wmma<1,1,0><<<g, 256, 0, stream>>>(x, w16, bias, y, Ci, Co, H, H,
                                                Ho, Ho, stride, pad, act, wbstride);
    return Ho;
  };
  auto castw = [&](const float* w, long long n, float scale) {
    k_w2bf<<<dim3((unsigned)((n + 255) / 256)), 256, 0, stream>>>(w, W16, n, scale);
  };
  auto blur = [&](const float* x, float* y, int C, int H, int up, int pad0,
                  int pad1, float gain, int addto) -> int {
    int D  = (H - 1) * up + 1;
    int Ho = D + pad0 + pad1 + up - 1 - 3;
    long long total = (long long)B * C * Ho * Ho;
    k_upfirdn<<<dim3((unsigned)((total + 255) / 256)), 256, 0, stream>>>(
        x, y, C, H, H, Ho, Ho, up, pad0, gain, addto);
    return Ho;
  };
  auto linear = [&](const float* x, const float* w, const float* b, float* y,
                    int K, int N, float scale, int act) {
    int tot = 8 * N;
    k_linear<<<dim3((tot + 255) / 256), 256, 0, stream>>>(x, w, b, y, K, N, scale, act);
  };

  // ================= Encoder =================
  castw(F(1), 128 * 12, 1.f / sqrtf(12.f));
  conv(F(0), W16, F(2), B0, 12, 128, 256, 1, 1, 0, 0, 1, 0);

  const int eci[6] = {128, 256, 512, 512, 512, 512};
  const int eco[6] = {256, 512, 512, 512, 512, 512};
  int Hc = 256;
  for (int i = 0; i < 6; ++i) {
    int ci = eci[i], co = eco[i];
    int base = 3 + 5 * i;
    float sc3 = 1.f / sqrtf((float)ci * 9.f);
    castw(F(base + 0), (long long)ci * ci * 9, sc3);
    conv(B0, W16, F(base + 1), B1, ci, ci, Hc, 3, 1, 1, 0, 1, 0);     // t1: HxH
    int Hb = blur(B1, B2, ci, Hc, 1, 2, 2, 1.f, 0);                   // H+1
    castw(F(base + 2), (long long)co * ci * 9, sc3);
    conv(B2, W16, F(base + 3), B1, ci, co, Hb, 3, 2, 0, 0, 1, 0);     // H/2
    Hb = blur(B0, B2, ci, Hc, 1, 1, 1, 1.f, 0);                       // H-1
    castw(F(base + 4), (long long)co * ci, 1.f / sqrtf((float)ci));
    conv(B2, W16, nullptr, B3, ci, co, Hb, 1, 2, 0, 0, 0, 0);         // skip: H/2
    Hc /= 2;
    long long tot = (long long)B * co * Hc * Hc;
    k_combine<<<dim3((unsigned)((tot + 255) / 256)), 256, 0, stream>>>(B1, B3, B0, tot);
  }
  // B0 = (8,512,4,4)
  castw(F(33), 512LL * 512 * 9, 1.f / sqrtf(512.f * 9.f));
  conv(B0, W16, F(34), FEAT, 512, 512, 4, 3, 1, 1, 0, 1, 0);
  linear(B0, F(35), F(36), HB, 8192, 512, 1.f / sqrtf(8192.f), 1);    // fin
  linear(HB, F(37), F(38), ZB, 512, 512, 1.f / sqrtf(512.f), 0);      // mu -> z

  // ================= Synthesis =================
  const float sstyle = 1.f / sqrtf(512.f);
  const int col[6] = {512, 512, 512, 512, 256, 128};
  float* Ocur = FEAT;
  float* Onext = B0;
  float* skip = nullptr;
  float* sknext = SK0;
  int ci2 = 512;
  int Hs = 4;
  for (int l = 0; l < 6; ++l) {
    int base = 41 + 14 * l;
    int co = col[l];
    // conv1 (up): style, modulate(+flip,+demod) -> bf16, dilated conv, blur, noise+bias+act
    linear(ZB, F(base + 1), F(base + 2), STY, 512, ci2, sstyle, 0);
    k_modulate<<<dim3(B * co), 256, 0, stream>>>(F(base + 0), STY, W16,
        co, ci2, 3, 3, 1.f / sqrtf((float)ci2 * 9.f), 1, 1);
    int Hu = conv(Ocur, W16, nullptr, B2, ci2, co, Hs, 3, 1, 0, 1, 0,
                  (long long)co * ci2 * 9);                        // 2H+1
    int Hb = blur(B2, B3, co, Hu, 1, 1, 1, 4.f, 0);                // 2H
    {
      long long tot = (long long)B * co * Hb * Hb;
      k_noise_bias_act<<<dim3((unsigned)((tot + 255) / 256)), 256, 0, stream>>>(
          B3, F(125 + 2 * l), F(base + 3), F(base + 4), co, Hb, Hb);
    }
    // conv2: style, modulate(+demod), 3x3 pad1, noise+bias+act
    linear(ZB, F(base + 6), F(base + 7), STY, 512, co, sstyle, 0);
    k_modulate<<<dim3(B * co), 256, 0, stream>>>(F(base + 5), STY, W16,
        co, co, 3, 3, 1.f / sqrtf((float)co * 9.f), 1, 0);
    conv(B3, W16, nullptr, Onext, co, co, Hb, 3, 1, 1, 0, 0,
         (long long)co * co * 9);
    {
      long long tot = (long long)B * co * Hb * Hb;
      k_noise_bias_act<<<dim3((unsigned)((tot + 255) / 256)), 256, 0, stream>>>(
          Onext, F(125 + 2 * l + 1), F(base + 8), F(base + 9), co, Hb, Hb);
    }
    // toRGB: style, modulate(no demod), 1x1 conv + bias; then skip upsample-add
    linear(ZB, F(base + 11), F(base + 12), STY, 512, co, sstyle, 0);
    k_modulate<<<dim3(B * 3), 256, 0, stream>>>(F(base + 10), STY, W16,
        3, co, 1, 1, 1.f / sqrtf((float)co), 0, 0);
    float* rgb = (l == 5) ? (float*)d_out : sknext;
    conv(Onext, W16, F(base + 13), rgb, co, 3, Hb, 1, 1, 0, 0, 0,
         (long long)3 * co);
    if (skip) blur(skip, rgb, 3, Hb / 2, 2, 2, 1, 4.f, 1);         // up=2 pad(2,1), add
    skip = rgb;
    sknext = (rgb == SK0) ? SK1 : SK0;
    Ocur = Onext;
    Onext = (Ocur == B0) ? B1 : B0;
    Hs = Hb;
    ci2 = co;
  }
}
